// AnchorTarget_5531917877592
// MI455X (gfx1250) — compile-verified, matrix operand order checked by reading, not run on hardware
//
#include <hip/hip_runtime.h>

// ---------------------------------------------------------------------------
// Anchor-target assignment for MI455X (gfx1250, wave32).
//   K0  init        : zero per-GT max accumulators
//   K1  k_iou       : inside flags, per-anchor max IoU, bbox targets,
//                     per-GT max (monotone-u32 atomicMax), GT table via TDM
//   K2  k_cand      : per-GT-argmax candidate mask + in-block scan
//   K3  k_scan256   : WMMA (16x16x32 f16) scan of 256 block sums
//   K4  k_fg        : fg mask (cand kept + high-IoU) + in-block scan
//   K5  k_scan256   : WMMA scan + bg quota (256 - min(n_fg,128))
//   K6  k_bg        : bg mask + in-block scan
//   K7  k_scan256   : WMMA scan
//   K8  k_labels    : final labels via global exclusive prefixes
// ---------------------------------------------------------------------------

#define NA       262144
#define RGT      100
#define BLK      1024
#define NBLK     (NA / BLK)     /* 256 */
#define POS_T    0.7f
#define NEG_T    0.3f
#define NUM_FG   128
#define NSAMPLE  256

typedef _Float16     v16h __attribute__((ext_vector_type(16)));
typedef float        v8f  __attribute__((ext_vector_type(8)));
typedef int          v4i  __attribute__((ext_vector_type(4)));
typedef int          v8i  __attribute__((ext_vector_type(8)));
typedef unsigned int v4u  __attribute__((ext_vector_type(4)));

// Monotone float <-> u32 mapping so unsigned atomicMax == float max.
__device__ __forceinline__ unsigned f32_mono(float f) {
  unsigned u = __float_as_uint(f);
  return ((int)u < 0) ? ~u : (u | 0x80000000u);
}
__device__ __forceinline__ float mono_f32(unsigned u) {
  return (u & 0x80000000u) ? __uint_as_float(u ^ 0x80000000u)
                           : __uint_as_float(~u);
}

// IoU with py-faster-rcnn +1 convention. Contraction off so that the two
// kernels that recompute this produce bit-identical values (needed for the
// iou == gt_max equality test).
__device__ __forceinline__ float iou_pair(float4 a, float4 g, float area_g) {
#pragma clang fp contract(off)
  float area_a = (a.z - a.x + 1.0f) * (a.w - a.y + 1.0f);
  float iw = fminf(a.z, g.z) - fmaxf(a.x, g.x) + 1.0f;
  float ih = fminf(a.w, g.w) - fmaxf(a.y, g.y) + 1.0f;
  float inter = fmaxf(iw, 0.0f) * fmaxf(ih, 0.0f);
  return inter / (area_a + area_g - inter);
}
__device__ __forceinline__ float gt_area(float4 g) {
#pragma clang fp contract(off)
  return (g.z - g.x + 1.0f) * (g.w - g.y + 1.0f);
}

// In-block (1024-wide) exclusive scan of a 0/1 value; stores per-element
// exclusive prefix and the block total.
__device__ __forceinline__ void block_scan_u32(unsigned v, unsigned* sc, int t,
                                               int i, unsigned* __restrict__ pre,
                                               unsigned* __restrict__ bsum) {
  sc[t] = v;
  __syncthreads();
  for (int off = 1; off < BLK; off <<= 1) {
    unsigned x = (t >= off) ? sc[t - off] : 0u;
    __syncthreads();
    sc[t] += x;
    __syncthreads();
  }
  pre[i] = sc[t] - v;
  if (t == BLK - 1) bsum[blockIdx.x] = sc[t];
}

__global__ void k_init(unsigned* gtmax) {
  if (threadIdx.x < 128) gtmax[threadIdx.x] = 0u;
}

// ---------------------------------------------------------------------------
__global__ __launch_bounds__(BLK) void k_iou(const float* __restrict__ anchors,
                                             const float* __restrict__ img,
                                             const float* __restrict__ gt,
                                             float* __restrict__ maxov,
                                             unsigned char* __restrict__ flags,
                                             unsigned* __restrict__ gtmax,
                                             float* __restrict__ targets) {
  __shared__ __align__(16) float s_gt[RGT * 4];
  __shared__ float s_area[RGT];
  __shared__ float4 s_anch[BLK];
  __shared__ unsigned s_gmax[128];
  const int t = threadIdx.x;
  const int i = blockIdx.x * BLK + t;

  __builtin_prefetch(anchors + (size_t)i * 4, 0, 1);   // global_prefetch_b8

  // Stage 100x4 f32 GT table into LDS.
#if defined(__HIP_DEVICE_COMPILE__) && __has_builtin(__builtin_amdgcn_tensor_load_to_lds)
  if (t == 0) {
    // TDM D#: 1D tile of 400 dwords, count=1, data_size=4B, type=2 (image).
    unsigned lds_off = (unsigned)(size_t)&s_gt[0];     // LDS byte offset = addr[31:0]
    unsigned long long ga = (unsigned long long)(size_t)gt;
    v4u g0 = { 1u,                                     // count=1 (valid user D#)
               lds_off,                                // lds_addr
               (unsigned)ga,                           // global_addr[31:0]
               ((unsigned)(ga >> 32) & 0x01FFFFFFu) | 0x80000000u }; // addr[56:32] | type=2
    v8i g1 = { 0x00020000,            // data_size=2 (4B), wg_mask=0 (not in cluster)
               (int)(400u << 16),     // tensor_dim0 = 400 (low16)
               (int)(1u << 16),       // tensor_dim1 = 1
               (int)(400u << 16),     // tile_dim0   = 400
               1,                     // tile_dim1 = 1, tile_dim2 = 0
               400,                   // tensor_dim0_stride = 400 (low32)
               (int)(400u << 16),     // tensor_dim1_stride = 400 (low16)
               0 };
    v4i z4 = { 0, 0, 0, 0 };
#if __clang_major__ >= 23
    v8i z8 = { 0, 0, 0, 0, 0, 0, 0, 0 };
    __builtin_amdgcn_tensor_load_to_lds(g0, g1, z4, z4, z8, 0);
#else
    __builtin_amdgcn_tensor_load_to_lds(g0, g1, z4, z4, 0);
#endif
    __builtin_amdgcn_s_wait_tensorcnt(0);
  }
#else
  if (t < RGT * 4) s_gt[t] = gt[t];
#endif
  __syncthreads();
  if (t < RGT) s_area[t] = gt_area(((const float4*)s_gt)[t]);
  if (t < 128) s_gmax[t] = 0u;
  float4 a = ((const float4*)anchors)[i];
  s_anch[t] = a;
  __syncthreads();

  const float W = img[1], H = img[0];
  const bool inside = (a.x >= 0.0f) & (a.y >= 0.0f) & (a.z < W) & (a.w < H);

  // Phase A: per-anchor max IoU over all GT.
  float mo = -1.0f;
  for (int r = 0; r < RGT; ++r) {
    float v = iou_pair(a, ((const float4*)s_gt)[r], s_area[r]);
    v = inside ? v : -1.0f;
    mo = fmaxf(mo, v);
  }
  maxov[i] = mo;
  flags[i] = inside ? 1u : 0u;

  // bbox targets: idx = clip(int(max_ov), 0, R-1)  (faithful to reference)
  int idx = (int)mo;
  idx = idx < 0 ? 0 : (idx > RGT - 1 ? RGT - 1 : idx);
  float4 g = ((const float4*)s_gt)[idx];
  float aw = a.z - a.x + 1.0f, ah = a.w - a.y + 1.0f;
  float acx = a.x + 0.5f * aw, acy = a.y + 0.5f * ah;
  float gw = g.z - g.x + 1.0f, gh = g.w - g.y + 1.0f;
  float gcx = g.x + 0.5f * gw, gcy = g.y + 0.5f * gh;
  float4 tq;
  tq.x = inside ? (gcx - acx) / aw : 0.0f;
  tq.y = inside ? (gcy - acy) / ah : 0.0f;
  tq.z = inside ? logf(gw / aw) : 0.0f;
  tq.w = inside ? logf(gh / ah) : 0.0f;
  ((float4*)targets)[i] = tq;

  // Phase B: per-GT max over this block's anchors (tiled recompute from LDS),
  // then one global atomic per GT per block.
  const int g_id = t & 127;
  const int chunk = t >> 7;                 // 8 chunks x 128 anchors
  if (g_id < RGT) {
    float4 gg = ((const float4*)s_gt)[g_id];
    float ag = s_area[g_id];
    unsigned best = 0u;
    for (int j = 0; j < BLK / 8; ++j) {
      float4 aa = s_anch[chunk * (BLK / 8) + j];
      bool ins = (aa.x >= 0.0f) & (aa.y >= 0.0f) & (aa.z < W) & (aa.w < H);
      float v = iou_pair(aa, gg, ag);
      v = ins ? v : -1.0f;
      unsigned u = f32_mono(v);
      best = best > u ? best : u;
    }
    atomicMax(&s_gmax[g_id], best);
  }
  __syncthreads();
  if (t < RGT) atomicMax(&gtmax[t], s_gmax[t]);
}

// ---------------------------------------------------------------------------
__global__ __launch_bounds__(BLK) void k_cand(const float* __restrict__ anchors,
                                              const float* __restrict__ img,
                                              const float* __restrict__ gt,
                                              const unsigned* __restrict__ gtmax,
                                              unsigned char* __restrict__ flags,
                                              unsigned* __restrict__ pre,
                                              unsigned* __restrict__ bsum) {
  __shared__ __align__(16) float s_gt[RGT * 4];
  __shared__ float s_area[RGT];
  __shared__ float s_gmax[RGT];
  __shared__ unsigned sc[BLK];
  const int t = threadIdx.x;
  const int i = blockIdx.x * BLK + t;
  if (t < RGT * 4) s_gt[t] = gt[t];
  __syncthreads();
  if (t < RGT) {
    s_area[t] = gt_area(((const float4*)s_gt)[t]);
    s_gmax[t] = mono_f32(gtmax[t]);
  }
  __syncthreads();

  float4 a = ((const float4*)anchors)[i];
  const float W = img[1], H = img[0];
  (void)W; (void)H;
  unsigned f = flags[i];
  const bool inside = (f & 1u) != 0u;
  bool cand = false;
  for (int r = 0; r < RGT; ++r) {
    float v = iou_pair(a, ((const float4*)s_gt)[r], s_area[r]);
    v = inside ? v : -1.0f;
    cand |= (v == s_gmax[r]);        // bit-exact recompute vs k_iou phase B
  }
  cand = cand && inside;
  flags[i] = (unsigned char)(f | (cand ? 2u : 0u));
  block_scan_u32(cand ? 1u : 0u, sc, t, i, pre, bsum);
}

// ---------------------------------------------------------------------------
// Scan of 256 block sums with one V_WMMA_F32_16X16X32_F16:
// X[k][n] = bsum[16n+k] (<=1024, exact f16). D = L x X gives the 16 inclusive
// column scans; column offsets close the scan. All sums < 2^24 -> exact f32.
__global__ __launch_bounds__(32) void k_scan256(const unsigned* __restrict__ bsum,
                                                unsigned* __restrict__ boff,
                                                unsigned* __restrict__ quotaOut) {
  __shared__ float X[16][16];
  __shared__ float INC[16][16];
  __shared__ float OFFC[16];
  const int l = (int)threadIdx.x;
  for (int e = l; e < 256; e += 32) X[e & 15][e >> 4] = (float)bsum[e];
  __syncthreads();

  const int mn = l & 15;                 // A: row m; B/D: col n
  const int hi8 = (l >= 16) ? 8 : 0;
  v16h a, b;
  for (int h = 0; h < 16; ++h) {
    // ISA 7.12.2 16-bit A 16x32 layout: lane<16 -> K {0..7,16..23},
    // lane>=16 -> K {8..15,24..31}; two halves per VGPR.
    int vv = h >> 1, pos = h & 1;
    int k = ((vv < 4) ? 2 * vv : 16 + 2 * (vv - 4)) + hi8 + pos;
    a[h] = (_Float16)((k < 16 && k <= mn) ? 1.0f : 0.0f);   // lower-tri ones
    b[h] = (k < 16) ? (_Float16)X[k][mn] : (_Float16)0.0f;  // mirrored layout
  }
  v8f c = {};
  c = __builtin_amdgcn_wmma_f32_16x16x32_f16(false, a, false, b,
                                             (short)0, c, false, false);
  for (int iv = 0; iv < 8; ++iv) INC[iv + hi8][mn] = c[iv];  // D layout
  __syncthreads();
  if (l < 16) {
    float s = 0.0f;
    for (int j = 0; j < l; ++j) s += INC[15][j];
    OFFC[l] = s;
  }
  __syncthreads();
  for (int e = l; e < 256; e += 32) {
    int k = e & 15, n = e >> 4;
    boff[e] = (unsigned)(INC[k][n] + OFFC[n] - X[k][n]);     // exclusive
  }
  if (quotaOut != nullptr && l == 0) {
    unsigned total = (unsigned)(INC[15][15] + OFFC[15]);
    unsigned nfg = total < (unsigned)NUM_FG ? total : (unsigned)NUM_FG;
    quotaOut[0] = (unsigned)NSAMPLE - nfg;
  }
}

// ---------------------------------------------------------------------------
__global__ __launch_bounds__(BLK) void k_fg(const float* __restrict__ maxov,
                                            unsigned char* __restrict__ flags,
                                            const unsigned* __restrict__ preA,
                                            const unsigned* __restrict__ boffA,
                                            unsigned* __restrict__ preB,
                                            unsigned* __restrict__ bsumB) {
  __shared__ unsigned sc[BLK];
  const int t = threadIdx.x;
  const int i = blockIdx.x * BLK + t;
  unsigned f = flags[i];
  bool cand_keep = (f & 2u) && (boffA[blockIdx.x] + preA[i] < (unsigned)RGT);
  bool pos = (f & 1u) && (maxov[i] >= POS_T);
  bool fg = cand_keep || pos;
  flags[i] = (unsigned char)(f | (fg ? 4u : 0u));
  block_scan_u32(fg ? 1u : 0u, sc, t, i, preB, bsumB);
}

__global__ __launch_bounds__(BLK) void k_bg(const float* __restrict__ maxov,
                                            unsigned char* __restrict__ flags,
                                            unsigned* __restrict__ preA,
                                            unsigned* __restrict__ bsumC) {
  __shared__ unsigned sc[BLK];
  const int t = threadIdx.x;
  const int i = blockIdx.x * BLK + t;
  unsigned f = flags[i];
  bool bg = (f & 1u) && (maxov[i] < NEG_T) && !(f & 4u);
  flags[i] = (unsigned char)(f | (bg ? 8u : 0u));
  block_scan_u32(bg ? 1u : 0u, sc, t, i, preA, bsumC);
}

__global__ __launch_bounds__(BLK) void k_labels(const unsigned char* __restrict__ flags,
                                                const unsigned* __restrict__ preB,
                                                const unsigned* __restrict__ boffB,
                                                const unsigned* __restrict__ preA,
                                                const unsigned* __restrict__ boffC,
                                                const unsigned* __restrict__ quota,
                                                float* __restrict__ labels) {
  const int t = threadIdx.x;
  const int i = blockIdx.x * BLK + t;
  unsigned f = flags[i];
  float lab = -1.0f;
  if ((f & 4u) && (boffB[blockIdx.x] + preB[i] < (unsigned)NUM_FG)) lab = 1.0f;
  else if ((f & 8u) && (boffC[blockIdx.x] + preA[i] < quota[0])) lab = 0.0f;
  labels[i] = lab;
}

// ---------------------------------------------------------------------------
extern "C" void kernel_launch(void* const* d_in, const int* in_sizes, int n_in,
                              void* d_out, int out_size, void* d_ws, size_t ws_size,
                              hipStream_t stream) {
  (void)in_sizes; (void)n_in; (void)out_size; (void)ws_size;
  const float* anchors = (const float*)d_in[0];   // (1,N,4) f32
  const float* img     = (const float*)d_in[1];   // (2,)    f32
  const float* gt      = (const float*)d_in[2];   // (1,R,4) f32
  float* labels  = (float*)d_out;                 // (1,N)
  float* targets = labels + NA;                   // (1,N,4)

  char* ws = (char*)d_ws;                         // ~3.42 MB used
  float*         maxov = (float*)ws;
  unsigned*      preA  = (unsigned*)(ws + 4ull * NA);
  unsigned*      preB  = (unsigned*)(ws + 8ull * NA);
  unsigned char* flags = (unsigned char*)(ws + 12ull * NA);
  unsigned*      gtmax = (unsigned*)(ws + 13ull * NA);
  unsigned* bsumA = gtmax + 128;
  unsigned* boffA = bsumA + 256;
  unsigned* bsumB = boffA + 256;
  unsigned* boffB = bsumB + 256;
  unsigned* bsumC = boffB + 256;
  unsigned* boffC = bsumC + 256;
  unsigned* quota = boffC + 256;

  hipLaunchKernelGGL(k_init, dim3(1), dim3(128), 0, stream, gtmax);
  hipLaunchKernelGGL(k_iou, dim3(NBLK), dim3(BLK), 0, stream,
                     anchors, img, gt, maxov, flags, gtmax, targets);
  hipLaunchKernelGGL(k_cand, dim3(NBLK), dim3(BLK), 0, stream,
                     anchors, img, gt, gtmax, flags, preA, bsumA);
  hipLaunchKernelGGL(k_scan256, dim3(1), dim3(32), 0, stream,
                     bsumA, boffA, (unsigned*)nullptr);
  hipLaunchKernelGGL(k_fg, dim3(NBLK), dim3(BLK), 0, stream,
                     maxov, flags, preA, boffA, preB, bsumB);
  hipLaunchKernelGGL(k_scan256, dim3(1), dim3(32), 0, stream,
                     bsumB, boffB, quota);
  hipLaunchKernelGGL(k_bg, dim3(NBLK), dim3(BLK), 0, stream,
                     maxov, flags, preA, bsumC);
  hipLaunchKernelGGL(k_scan256, dim3(1), dim3(32), 0, stream,
                     bsumC, boffC, (unsigned*)nullptr);
  hipLaunchKernelGGL(k_labels, dim3(NBLK), dim3(BLK), 0, stream,
                     flags, preB, boffB, preA, boffC, quota, labels);
}
